// SwinTransformerBlock_83356725281345
// MI455X (gfx1250) — compile-verified, matrix operand order checked by reading, not run on hardware
//
#include <hip/hip_runtime.h>
#include <hip/hip_fp16.h>
#include <math.h>

typedef __attribute__((ext_vector_type(16))) _Float16 v16h;
typedef __attribute__((ext_vector_type(8)))  _Float16 v8h;
typedef __attribute__((ext_vector_type(8)))  float    v8f;
typedef __attribute__((ext_vector_type(4)))  unsigned int u32x4;
typedef __attribute__((ext_vector_type(8)))  int      i32x8;
typedef __attribute__((ext_vector_type(4)))  int      i32x4;

#define DIM     192
#define HEADS   6
#define HD      32
#define WIN     8
#define SHIFT   4
#define GROUPS  8
#define MLP_HID 768
#define HW      65536
#define NTOK    64
#define EPS     1e-5f

// ---------------- WMMA fragment helpers (wave32, 16x16x32 f16) ----------------
// A (M x K, row-major) and B stored N-major (row = output column, K contiguous):
// lane L -> row = base_row + (L&15); K packing: VGPR i<4 holds K = kb+2i,kb+2i+1
// with kb = (L<16) ? 0 : 8; VGPR 4..7 hold K = 16+kb .. 16+kb+7.
// => halves [0..7] = row[k0+kb .. +7], halves [8..15] = row[k0+16+kb .. +7].
__device__ __forceinline__ v16h frag_rowmajor(const _Float16* base, int pitch,
                                              int row0, int k0, int lane) {
    int r  = row0 + (lane & 15);
    int kb = (lane >> 4) ? 8 : 0;
    const _Float16* p = base + (size_t)r * pitch + k0 + kb;
    v16h f;
    *(v8h*)&f         = *(const v8h*)(p);
    *(((v8h*)&f) + 1) = *(const v8h*)(p + 16);
    return f;
}

__device__ __forceinline__ v8f wmma16(v16h a, v16h b, v8f c) {
    return __builtin_amdgcn_wmma_f32_16x16x32_f16(false, a, false, b,
                                                  (short)0, c, false, false);
}

// ---------------- weight convert: fp32 -> f16, transposed to N-major ----------
__global__ void wconv_kernel(const float* __restrict__ qkv_w,
                             const float* __restrict__ proj_w,
                             const float* __restrict__ mlp_w1,
                             const float* __restrict__ mlp_w2,
                             _Float16* __restrict__ qkvWt,
                             _Float16* __restrict__ projWt,
                             _Float16* __restrict__ w1t,
                             _Float16* __restrict__ w2t) {
    int i = blockIdx.x * 256 + threadIdx.x;
    if (i < DIM * 3 * DIM) { int k = i / (3*DIM), n = i % (3*DIM); qkvWt[n*DIM + k] = (_Float16)qkv_w[i]; }
    if (i < DIM * DIM)     { int k = i / DIM,     n = i % DIM;     projWt[n*DIM + k] = (_Float16)proj_w[i]; }
    if (i < DIM * MLP_HID) { int k = i / MLP_HID, n = i % MLP_HID; w1t[n*DIM + k]    = (_Float16)mlp_w1[i]; }
    if (i < MLP_HID * DIM) { int k = i / DIM,     n = i % DIM;     w2t[n*MLP_HID + k] = (_Float16)mlp_w2[i]; }
}

__global__ void zero_stats_kernel(float* __restrict__ stats) {
    if (threadIdx.x < 128) stats[threadIdx.x] = 0.0f;
}

// ---------------- GroupNorm reduction: one block per (b, c) plane -------------
__global__ __launch_bounds__(256) void gn_reduce_kernel(const float* __restrict__ src,
                                                        float* __restrict__ stats) {
    __shared__ float ssum[256], ssq[256];
    int bc = blockIdx.x;                 // 0 .. 4*192-1
    int b = bc / DIM, c = bc % DIM;
    const float* p = src + ((size_t)bc << 16);
    float s = 0.f, q = 0.f;
    for (int i = threadIdx.x; i < HW; i += 256) {
        float v = p[i]; s += v; q += v * v;
    }
    ssum[threadIdx.x] = s; ssq[threadIdx.x] = q;
    __syncthreads();
    for (int st = 128; st > 0; st >>= 1) {
        if (threadIdx.x < st) {
            ssum[threadIdx.x] += ssum[threadIdx.x + st];
            ssq[threadIdx.x]  += ssq[threadIdx.x + st];
        }
        __syncthreads();
    }
    if (threadIdx.x == 0) {
        int g = c / (DIM / GROUPS);
        atomicAdd(&stats[(b * GROUPS + g) * 2 + 0], ssum[0]);
        atomicAdd(&stats[(b * GROUPS + g) * 2 + 1], ssq[0]);
    }
}

// ---------------- fused GN1 + shifted-window attention + proj + residual ------
__global__ __launch_bounds__(256) void attn_kernel(
    const float* __restrict__ x, const _Float16* __restrict__ qkvWt,
    const float* __restrict__ qkv_b, const _Float16* __restrict__ projWt,
    const float* __restrict__ proj_b, const float* __restrict__ n1w,
    const float* __restrict__ n1b, const float* __restrict__ stats,
    float* __restrict__ x1) {
    extern __shared__ char smem[];
    _Float16* lds_x   = (_Float16*)smem;                        // 64 x 192 (later: Vt 192 x 64)
    _Float16* lds_qkv = lds_x + NTOK * DIM;                     // 64 x 576
    float*    lds_S   = (float*)(lds_qkv + NTOK * 3 * DIM);     // 6 x 64 x 64
    _Float16* lds_P   = (_Float16*)(lds_S + HEADS*NTOK*NTOK);   // 6 x 64 x 64
    _Float16* lds_o   = lds_P + HEADS * NTOK * NTOK;            // 64 x 192

    const int wgid = blockIdx.x;
    const int b   = wgid >> 10;
    const int wy0 = (wgid >> 5) & 31;
    const int wx0 = wgid & 31;
    const int tid = threadIdx.x;
    const int wave = tid >> 5, lane = tid & 31;
    const float inv_cnt = 1.0f / (float)((DIM / GROUPS) * HW);

    // gather window tokens (shift-roll folded into index), apply GN1, cast f16
    for (int idx = tid; idx < NTOK * DIM; idx += 256) {
        int c = idx >> 6, n = idx & 63;
        int h = ((wy0 << 3) + (n >> 3) + SHIFT) & 255;
        int w = ((wx0 << 3) + (n & 7) + SHIFT) & 255;
        int g = c / (DIM / GROUPS);
        float mean = stats[(b * GROUPS + g) * 2 + 0] * inv_cnt;
        float var  = stats[(b * GROUPS + g) * 2 + 1] * inv_cnt - mean * mean;
        float v = x[((size_t)(b * DIM + c) << 16) + (h << 8) + w];
        v = (v - mean) * rsqrtf(var + EPS) * n1w[c] + n1b[c];
        lds_x[n * DIM + c] = (_Float16)v;
    }
    __syncthreads();

    // QKV: (64x192) @ (192x576); 4 x 36 tiles, K chunks of 32
    for (int t = wave; t < 4 * 36; t += 8) {
        int mi = t / 36, ni = t % 36;
        v8f acc = {};
#pragma unroll
        for (int kc = 0; kc < 6; ++kc) {
            v16h a  = frag_rowmajor(lds_x, DIM, mi * 16, kc * 32, lane);
            v16h bb = frag_rowmajor(qkvWt, DIM, ni * 16, kc * 32, lane);
            acc = wmma16(a, bb, acc);
        }
        int n = ni * 16 + (lane & 15);
        float bias = qkv_b[n];
        int mb = (lane >> 4) ? 8 : 0;
#pragma unroll
        for (int r = 0; r < 8; ++r)
            lds_qkv[(mi * 16 + mb + r) * (3 * DIM) + n] = (_Float16)(acc[r] + bias);
    }
    __syncthreads();

    // transpose V into Vt[192][64] (reuses lds_x, dead after QKV) so the O-phase
    // B fragments become contiguous b128 LDS reads instead of 16-wide gathers
    _Float16* vt = lds_x;
    for (int idx = tid; idx < NTOK * DIM; idx += 256) {
        int d = idx >> 6, n = idx & 63;
        vt[d * NTOK + n] = lds_qkv[n * (3 * DIM) + 2 * DIM + d];
    }

    // S = scale * Q K^T per head (K dim = hd = 32 -> one WMMA per tile)
    if (wave < HEADS) {
        const _Float16* qb  = lds_qkv + wave * HD;
        const _Float16* kbp = lds_qkv + DIM + wave * HD;
        float* S = lds_S + wave * NTOK * NTOK;
        const float scale = 0.17677669529663687f;  // 1/sqrt(32)
#pragma unroll
        for (int mi = 0; mi < 4; ++mi)
#pragma unroll
            for (int ni = 0; ni < 4; ++ni) {
                v8f acc = {};
                v16h a  = frag_rowmajor(qb,  3 * DIM, mi * 16, 0, lane);
                v16h bb = frag_rowmajor(kbp, 3 * DIM, ni * 16, 0, lane);
                acc = wmma16(a, bb, acc);
                int n = ni * 16 + (lane & 15);
                int mb = (lane >> 4) ? 8 : 0;
#pragma unroll
                for (int r = 0; r < 8; ++r)
                    S[(mi * 16 + mb + r) * NTOK + n] = acc[r] * scale;
            }
    }
    __syncthreads();

    // softmax rows (6*64 rows over 256 threads), emit P as f16
    for (int rr = tid; rr < HEADS * NTOK; rr += 256) {
        float* S = lds_S + rr * NTOK;
        _Float16* P = lds_P + rr * NTOK;
        float mx = -1e30f;
        for (int j = 0; j < NTOK; ++j) mx = fmaxf(mx, S[j]);
        float sm = 0.f;
        for (int j = 0; j < NTOK; ++j) { float e = __expf(S[j] - mx); S[j] = e; sm += e; }
        float inv = 1.0f / sm;
        for (int j = 0; j < NTOK; ++j) P[j] = (_Float16)(S[j] * inv);
    }
    __syncthreads();

    // O = P @ V per head (B from Vt, row-major b128 fragments)
    if (wave < HEADS) {
        const _Float16* P = lds_P + wave * NTOK * NTOK;
#pragma unroll
        for (int mi = 0; mi < 4; ++mi)
#pragma unroll
            for (int nj = 0; nj < 2; ++nj) {
                v8f acc = {};
#pragma unroll
                for (int kc = 0; kc < 2; ++kc) {
                    v16h a  = frag_rowmajor(P, NTOK, mi * 16, kc * 32, lane);
                    v16h bb = frag_rowmajor(vt, NTOK, wave * HD + nj * 16, kc * 32, lane);
                    acc = wmma16(a, bb, acc);
                }
                int n = nj * 16 + (lane & 15);
                int mb = (lane >> 4) ? 8 : 0;
#pragma unroll
                for (int r = 0; r < 8; ++r)
                    lds_o[(mi * 16 + mb + r) * DIM + wave * HD + n] = (_Float16)acc[r];
            }
    }
    __syncthreads();

    // proj + reverse-roll scatter + residual
    for (int t = wave; t < 4 * 12; t += 8) {
        int mi = t / 12, ni = t % 12;
        v8f acc = {};
#pragma unroll
        for (int kc = 0; kc < 6; ++kc) {
            v16h a  = frag_rowmajor(lds_o, DIM, mi * 16, kc * 32, lane);
            v16h bb = frag_rowmajor(projWt, DIM, ni * 16, kc * 32, lane);
            acc = wmma16(a, bb, acc);
        }
        int c = ni * 16 + (lane & 15);
        float bias = proj_b[c];
        int mb = (lane >> 4) ? 8 : 0;
#pragma unroll
        for (int r = 0; r < 8; ++r) {
            int n = mi * 16 + mb + r;
            int hh = ((wy0 << 3) + (n >> 3) + SHIFT) & 255;
            int ww = ((wx0 << 3) + (n & 7) + SHIFT) & 255;
            size_t off = ((size_t)(b * DIM + c) << 16) + (hh << 8) + ww;
            x1[off] = x[off] + acc[r] + bias;
        }
    }
}

// ---------------- fused GN2 + MLP + residual ----------------------------------
// The activation tile (64 contiguous pixels x 192 channels, channel stride
// 65536) is an affine 2D tile: fetched via the Tensor Data Mover into LDS
// (stage, fp32, at dynamic-LDS offset 0), tracked with TENSORcnt. The staged
// tile also supplies the fc2 residual (removes one global pass over x1).
// TDM call is UNCONDITIONAL this round to confirm the builtin lowers.
__global__ __launch_bounds__(256) void mlp_kernel(
    const float* __restrict__ x1, const _Float16* __restrict__ w1t,
    const float* __restrict__ b1, const _Float16* __restrict__ w2t,
    const float* __restrict__ b2, const float* __restrict__ n2w,
    const float* __restrict__ n2b, const float* __restrict__ stats,
    float* __restrict__ out) {
    extern __shared__ char smem[];
    float*    stage = (float*)smem;                       // 192 x 64 fp32 (TDM dest, offset 0)
    _Float16* lds_x = (_Float16*)(smem + NTOK * DIM * 4); // 64 x 192 f16
    _Float16* lds_h = lds_x + NTOK * DIM;                 // 64 x 768 f16

    const int t0  = blockIdx.x * NTOK;
    const int b   = t0 >> 16;
    const int hw0 = t0 & 65535;
    const int tid = threadIdx.x, wave = tid >> 5, lane = tid & 31;
    const float inv_cnt = 1.0f / (float)((DIM / GROUPS) * HW);

    if (wave == 0) {
        // D# group 0: count=1 | lds_addr=0 | global_addr | type=2
        unsigned long long ga = (unsigned long long)(size_t)x1 +
                                ((unsigned long long)b * DIM * HW + (unsigned long long)hw0) * 4ull;
        u32x4 g0 = { 1u, 0u, (unsigned)ga, (unsigned)(ga >> 32) | 0x80000000u };
        // D# group 1: data_size=4B; tensor_dim0=65536 (stride 65536),
        // tensor_dim1=768 (stride 65536); tile_dim0=64, tile_dim1=192
        i32x8 g1 = { 0x00020000, 0, 0x03000001, 0x00400000,
                     0x000000C0, 0x00010000, 0, 1 };
        i32x4 gz = { 0, 0, 0, 0 };
#if defined(__clang_major__) && (__clang_major__ >= 23)
        i32x8 gz8 = { 0, 0, 0, 0, 0, 0, 0, 0 };
        __builtin_amdgcn_tensor_load_to_lds(g0, g1, gz, gz, gz8, 0);
#else
        __builtin_amdgcn_tensor_load_to_lds(g0, g1, gz, gz, 0);
#endif
        __builtin_amdgcn_s_wait_tensorcnt(0);
    }
    __syncthreads();

    // GN2 normalize from staged tile, cast f16 (token-major for A fragments)
    for (int idx = tid; idx < NTOK * DIM; idx += 256) {
        int c = idx >> 6, n = idx & 63;
        int g = c / (DIM / GROUPS);
        float mean = stats[(b * GROUPS + g) * 2 + 0] * inv_cnt;
        float var  = stats[(b * GROUPS + g) * 2 + 1] * inv_cnt - mean * mean;
        float v = stage[c * NTOK + n];
        v = (v - mean) * rsqrtf(var + EPS) * n2w[c] + n2b[c];
        lds_x[n * DIM + c] = (_Float16)v;
    }
    __syncthreads();

    // fc1 + exact GELU
    for (int t = wave; t < 4 * 48; t += 8) {
        int mi = t / 48, ni = t % 48;
        v8f acc = {};
#pragma unroll
        for (int kc = 0; kc < 6; ++kc) {
            v16h a  = frag_rowmajor(lds_x, DIM, mi * 16, kc * 32, lane);
            v16h bb = frag_rowmajor(w1t, DIM, ni * 16, kc * 32, lane);
            acc = wmma16(a, bb, acc);
        }
        int j = ni * 16 + (lane & 15);
        float bias = b1[j];
        int mb = (lane >> 4) ? 8 : 0;
#pragma unroll
        for (int r = 0; r < 8; ++r) {
            float v = acc[r] + bias;
            v = 0.5f * v * (1.0f + erff(v * 0.70710678118654752f));
            lds_h[(mi * 16 + mb + r) * MLP_HID + j] = (_Float16)v;
        }
    }
    __syncthreads();

    // fc2 + residual (residual read from staged LDS tile, not global)
    for (int t = wave; t < 4 * 12; t += 8) {
        int mi = t / 12, ni = t % 12;
        v8f acc = {};
#pragma unroll
        for (int kc = 0; kc < 24; ++kc) {
            v16h a  = frag_rowmajor(lds_h, MLP_HID, mi * 16, kc * 32, lane);
            v16h bb = frag_rowmajor(w2t, MLP_HID, ni * 16, kc * 32, lane);
            acc = wmma16(a, bb, acc);
        }
        int c = ni * 16 + (lane & 15);
        float bias = b2[c];
        int mb = (lane >> 4) ? 8 : 0;
#pragma unroll
        for (int r = 0; r < 8; ++r) {
            int n = mi * 16 + mb + r;
            size_t off = ((size_t)(b * DIM + c) << 16) + hw0 + n;
            out[off] = stage[c * NTOK + n] + acc[r] + bias;
        }
    }
}

// ---------------- host launch --------------------------------------------------
extern "C" void kernel_launch(void* const* d_in, const int* in_sizes, int n_in,
                              void* d_out, int out_size, void* d_ws, size_t ws_size,
                              hipStream_t stream) {
    (void)in_sizes; (void)n_in; (void)out_size; (void)ws_size;
    const float* x      = (const float*)d_in[0];
    const float* qkv_w  = (const float*)d_in[1];
    const float* qkv_b  = (const float*)d_in[2];
    const float* proj_w = (const float*)d_in[3];
    const float* proj_b = (const float*)d_in[4];
    const float* n1w    = (const float*)d_in[5];
    const float* n1b    = (const float*)d_in[6];
    const float* n2w    = (const float*)d_in[7];
    const float* n2b    = (const float*)d_in[8];
    const float* mlp_w1 = (const float*)d_in[9];
    const float* mlp_b1 = (const float*)d_in[10];
    const float* mlp_w2 = (const float*)d_in[11];
    const float* mlp_b2 = (const float*)d_in[12];
    float* out = (float*)d_out;

    // workspace carve-up (256B aligned)
    char* ws = (char*)d_ws;
    size_t off = 0;
    auto carve = [&](size_t bytes) { char* p = ws + off; off = (off + bytes + 255) & ~(size_t)255; return p; };
    _Float16* qkvWt  = (_Float16*)carve((size_t)(3 * DIM) * DIM * 2);
    _Float16* projWt = (_Float16*)carve((size_t)DIM * DIM * 2);
    _Float16* w1t    = (_Float16*)carve((size_t)MLP_HID * DIM * 2);
    _Float16* w2t    = (_Float16*)carve((size_t)DIM * MLP_HID * 2);
    float*    stats  = (float*)carve(128 * 4);          // [0..63]=GN1, [64..127]=GN2
    float*    stats1 = stats;
    float*    stats2 = stats + 64;
    float*    x1     = (float*)carve((size_t)4 * DIM * HW * 4);

    wconv_kernel<<<(DIM * MLP_HID + 255) / 256, 256, 0, stream>>>(
        qkv_w, proj_w, mlp_w1, mlp_w2, qkvWt, projWt, w1t, w2t);
    zero_stats_kernel<<<1, 128, 0, stream>>>(stats);
    gn_reduce_kernel<<<4 * DIM, 256, 0, stream>>>(x, stats1);

    size_t attn_lds = (size_t)NTOK * DIM * 2 + (size_t)NTOK * 3 * DIM * 2 +
                      (size_t)HEADS * NTOK * NTOK * 4 + (size_t)HEADS * NTOK * NTOK * 2 +
                      (size_t)NTOK * DIM * 2;  // 270336 B
    attn_kernel<<<4096, 256, attn_lds, stream>>>(x, qkvWt, qkv_b, projWt, proj_b,
                                                 n1w, n1b, stats1, x1);

    gn_reduce_kernel<<<4 * DIM, 256, 0, stream>>>(x1, stats2);

    size_t mlp_lds = (size_t)NTOK * DIM * 4 + (size_t)NTOK * DIM * 2 +
                     (size_t)NTOK * MLP_HID * 2;  // 172032 B
    mlp_kernel<<<4096, 256, mlp_lds, stream>>>(x1, w1t, mlp_b1, w2t, mlp_b2,
                                               n2w, n2b, stats2, out);
}
